// TAMCaD_31353261260855
// MI455X (gfx1250) — matrix operand order repaired and checked by reading, not compiled, not run on hardware
//
#include <hip/hip_runtime.h>

// ---------------------------------------------------------------------------
// TAMCaD temporal attention for MI455X (gfx1250, wave32, WMMA).
//
// Roofline: ~17.2 GFLOP vs ~1.5 GB of HBM traffic -> memory bound (~66us at
// 23.3 TB/s). Strategy: coalesced f32 loads along t, convert to bf16 into LDS,
// run per-t 32x64x128 / 32x128x64 GEMMs with v_wmma_f32_16x16x32_bf16
// (f32 accumulate), softmax in registers with 16-lane shfl_xor reductions.
// ---------------------------------------------------------------------------

#define B_    4
#define G_    32     // query groups
#define GKV   64     // kv groups (k concat k_instant)
#define HID_  512
#define H_    4      // heads
#define DK    128    // head dim
#define T_    1024
#define TT    8      // t-values per workgroup (one wave each)
#define DKP   132    // padded LDS row pitch for d (bank-conflict hedge)
#define GKVP  68     // padded LDS row pitch for kv
#define NTHREADS 256 // 8 waves of 32
#define SCALE 0.08838834764831845f   // 1/sqrt(128), TEMPERATURE=1

typedef __attribute__((ext_vector_type(16))) __bf16 v16bf;
typedef __attribute__((ext_vector_type(8)))  float  v8f;

static __device__ __forceinline__ v8f wmma_bf16(v16bf a, v16bf b, v8f c) {
  // D = A(16x32 bf16) x B(32x16 bf16) + C(16x16 f32)
  return __builtin_amdgcn_wmma_f32_16x16x32_bf16(
      false, a, false, b, (short)0, c, false, false);
}

// A fragment (16x32 bf16), source stored row-major M x K with pitch `pitch`.
// Layout: lanes 0-15 -> M=lane, VGPR v holds K = 2v(+1);
// lanes 16-31 same M, K offset +16.
static __device__ __forceinline__ v16bf load_a_frag(const __bf16* base, int pitch) {
  const int lane = threadIdx.x & 31;
  const int half = lane >> 4;
  const int m    = lane & 15;
  v16bf a;
#pragma unroll
  for (int v = 0; v < 8; ++v) {
    int kb = 2 * v + 16 * half;
    a[2 * v]     = base[m * pitch + kb];
    a[2 * v + 1] = base[m * pitch + kb + 1];
  }
  return a;
}

// B fragment (32x16 bf16) where LDS stores the operand N-major: Mx[N][K].
// Same contiguous k order as load_a_frag: lanes 0-15 hold K=0..15, lanes 16-31 K=16..31.
static __device__ __forceinline__ v16bf load_b_nmajor(const __bf16* base, int pitch) {
  const int lane = threadIdx.x & 31;
  const int half = lane >> 4;
  const int n    = lane & 15;
  v16bf b;
#pragma unroll
  for (int v = 0; v < 8; ++v) {
    int kb = 2 * v + 16 * half;
    b[2 * v]     = base[n * pitch + kb];
    b[2 * v + 1] = base[n * pitch + kb + 1];
  }
  return b;
}

// B fragment (32x16 bf16) where LDS stores the operand K-major: Mx[K][N].
static __device__ __forceinline__ v16bf load_b_kmajor(const __bf16* base, int pitch) {
  const int lane = threadIdx.x & 31;
  const int half = lane >> 4;
  const int n    = lane & 15;
  v16bf b;
#pragma unroll
  for (int v = 0; v < 8; ++v) {
    int kb = 2 * v + 16 * half;
    b[2 * v]     = base[kb * pitch + n];
    b[2 * v + 1] = base[(kb + 1) * pitch + n];
  }
  return b;
}

__global__ void zero_f32(float* p, int n) {
  int i = blockIdx.x * blockDim.x + threadIdx.x;
  if (i < n) p[i] = 0.0f;
}

__global__ __launch_bounds__(NTHREADS)
void tamcad_attn(const float* __restrict__ q,
                 const float* __restrict__ kk,
                 const float* __restrict__ vv,
                 const float* __restrict__ ki,
                 const float* __restrict__ vi,
                 float* __restrict__ xout,
                 float* __restrict__ amean) {
  // LDS: phase A = Q + K (bf16). Phase B reuses sQ for bf16 att, sK for V.
  __shared__ __bf16 sQ[TT * G_  * DKP];  //  67,584 B
  __shared__ __bf16 sK[TT * GKV * DKP];  // 135,168 B  (total ~203 KB < 320 KB)

  const int nTB = T_ / TT;
  const int gid = blockIdx.x;
  const int tb  = gid % nTB;
  const int h   = (gid / nTB) % H_;
  const int b   = gid / (nTB * H_);
  const int t0  = tb * TT;

  const int tid  = threadIdx.x;
  const int w    = tid >> 5;        // wave id == local t index
  const int lane = tid & 31;
  const int half = lane >> 4;
  const int lcol = lane & 15;
  const int t    = t0 + w;

  // ---- Stage Q (32x128) and K (64x128) for 8 t's, f32 -> bf16 -------------
  for (int row = tid; row < G_ * DK; row += NTHREADS) {
    const int g = row >> 7, dd = row & 127;
    const float4* src = reinterpret_cast<const float4*>(
        q + (size_t)((b * G_ + g) * HID_ + h * DK + dd) * T_ + t0);
    float4 p0 = src[0], p1 = src[1];
    float vals[8] = {p0.x, p0.y, p0.z, p0.w, p1.x, p1.y, p1.z, p1.w};
#pragma unroll
    for (int tt = 0; tt < TT; ++tt)
      sQ[(tt * G_ + g) * DKP + dd] = (__bf16)vals[tt];
  }
  for (int row = tid; row < GKV * DK; row += NTHREADS) {
    const int kg = row >> 7, dd = row & 127;
    const float* sp = (kg < G_)
        ? (kk + (size_t)((b * G_ + kg)        * HID_ + h * DK + dd) * T_ + t0)
        : (ki + (size_t)((b * G_ + (kg - G_)) * HID_ + h * DK + dd) * T_ + t0);
    const float4* src = reinterpret_cast<const float4*>(sp);
    float4 p0 = src[0], p1 = src[1];
    float vals[8] = {p0.x, p0.y, p0.z, p0.w, p1.x, p1.y, p1.z, p1.w};
#pragma unroll
    for (int tt = 0; tt < TT; ++tt)
      sK[(tt * GKV + kg) * DKP + dd] = (__bf16)vals[tt];
  }
  __syncthreads();

  const __bf16* qT = &sQ[w * G_  * DKP];
  const __bf16* kT = &sK[w * GKV * DKP];

  // ---- Logits: (32x128) x (128x64) via 2x4 tiles, K chunks of 32 ----------
  v8f acc[2][4];
#pragma unroll
  for (int gt = 0; gt < 2; ++gt)
#pragma unroll
    for (int kt = 0; kt < 4; ++kt) acc[gt][kt] = (v8f){0,0,0,0,0,0,0,0};

#pragma unroll
  for (int dc = 0; dc < 4; ++dc) {
    const int db = dc * 32;
    v16bf aq0 = load_a_frag(&qT[(0  * DKP) + db], DKP);
    v16bf aq1 = load_a_frag(&qT[(16 * DKP) + db], DKP);
#pragma unroll
    for (int kt = 0; kt < 4; ++kt) {
      v16bf bk = load_b_nmajor(&kT[(kt * 16) * DKP + db], DKP);
      acc[0][kt] = wmma_bf16(aq0, bk, acc[0][kt]);
      acc[1][kt] = wmma_bf16(aq1, bk, acc[1][kt]);
    }
  }

  // ---- Softmax over kv=64 (4 accumulators x 16 lanes within each half) ----
  // C layout: lane<16 -> row gt*16+r, lane>=16 -> row gt*16+r+8, col = kt*16+lcol.
  __bf16* attW = &sQ[w * G_ * DKP];  // reuse own Q slice for bf16 att (32x64)
#pragma unroll
  for (int gt = 0; gt < 2; ++gt) {
#pragma unroll
    for (int r = 0; r < 8; ++r) {
      float x0 = acc[gt][0][r] * SCALE;
      float x1 = acc[gt][1][r] * SCALE;
      float x2 = acc[gt][2][r] * SCALE;
      float x3 = acc[gt][3][r] * SCALE;
      float m = fmaxf(fmaxf(x0, x1), fmaxf(x2, x3));
      m = fmaxf(m, __shfl_xor(m, 1, 32));
      m = fmaxf(m, __shfl_xor(m, 2, 32));
      m = fmaxf(m, __shfl_xor(m, 4, 32));
      m = fmaxf(m, __shfl_xor(m, 8, 32));
      float e0 = __expf(x0 - m), e1 = __expf(x1 - m);
      float e2 = __expf(x2 - m), e3 = __expf(x3 - m);
      float s = e0 + e1 + e2 + e3;
      s += __shfl_xor(s, 1, 32);
      s += __shfl_xor(s, 2, 32);
      s += __shfl_xor(s, 4, 32);
      s += __shfl_xor(s, 8, 32);
      const float inv = 1.0f / s;
      const int g = gt * 16 + (half ? (r + 8) : r);
      float a0 = e0 * inv, a1 = e1 * inv, a2 = e2 * inv, a3 = e3 * inv;
      attW[g * GKVP + 0  + lcol] = (__bf16)a0;
      attW[g * GKVP + 16 + lcol] = (__bf16)a1;
      attW[g * GKVP + 32 + lcol] = (__bf16)a2;
      attW[g * GKVP + 48 + lcol] = (__bf16)a3;
      float* am = amean + ((size_t)(b * G_ + g) * GKV) * T_ + t;
      atomicAdd(am + (size_t)(0  + lcol) * T_, 0.25f * a0);
      atomicAdd(am + (size_t)(16 + lcol) * T_, 0.25f * a1);
      atomicAdd(am + (size_t)(32 + lcol) * T_, 0.25f * a2);
      atomicAdd(am + (size_t)(48 + lcol) * T_, 0.25f * a3);
    }
  }
  __syncthreads();   // all waves done reading K before V overwrites sK

  // ---- Stage V (64x128) for 8 t's into sK ---------------------------------
  for (int row = tid; row < GKV * DK; row += NTHREADS) {
    const int kg = row >> 7, dd = row & 127;
    const float* sp = (kg < G_)
        ? (vv + (size_t)((b * G_ + kg)        * HID_ + h * DK + dd) * T_ + t0)
        : (vi + (size_t)((b * G_ + (kg - G_)) * HID_ + h * DK + dd) * T_ + t0);
    const float4* src = reinterpret_cast<const float4*>(sp);
    float4 p0 = src[0], p1 = src[1];
    float vals[8] = {p0.x, p0.y, p0.z, p0.w, p1.x, p1.y, p1.z, p1.w};
#pragma unroll
    for (int tt = 0; tt < TT; ++tt)
      sK[(tt * GKV + kg) * DKP + dd] = (__bf16)vals[tt];
  }
  __syncthreads();

  // ---- x = att (32x64) x V (64x128), tiles 2(g) x 8(d), K chunks of 32 ----
  const __bf16* vT = &sK[w * GKV * DKP];
#pragma unroll
  for (int gt = 0; gt < 2; ++gt) {
    v16bf a0 = load_a_frag(&attW[(gt * 16) * GKVP + 0],  GKVP);
    v16bf a1 = load_a_frag(&attW[(gt * 16) * GKVP + 32], GKVP);
#pragma unroll
    for (int dt = 0; dt < 8; ++dt) {
      v8f c = (v8f){0,0,0,0,0,0,0,0};
      v16bf b0 = load_b_kmajor(&vT[0  * DKP + dt * 16], DKP);  // kv 0..31
      v16bf b1 = load_b_kmajor(&vT[32 * DKP + dt * 16], DKP);  // kv 32..63
      c = wmma_bf16(a0, b0, c);
      c = wmma_bf16(a1, b1, c);
      const int d = dt * 16 + lcol;
#pragma unroll
      for (int r = 0; r < 8; ++r) {
        const int g = gt * 16 + (half ? (r + 8) : r);
        xout[(size_t)((b * G_ + g) * HID_ + h * DK + d) * T_ + t] = c[r];
      }
    }
  }
}

extern "C" void kernel_launch(void* const* d_in, const int* in_sizes, int n_in,
                              void* d_out, int out_size, void* d_ws, size_t ws_size,
                              hipStream_t stream) {
  (void)in_sizes; (void)n_in; (void)d_ws; (void)ws_size; (void)out_size;
  const float* q  = (const float*)d_in[0];
  const float* k  = (const float*)d_in[1];
  const float* v  = (const float*)d_in[2];
  const float* ki = (const float*)d_in[3];
  const float* vi = (const float*)d_in[4];
  float* xout  = (float*)d_out;
  float* amean = xout + (size_t)B_ * G_ * HID_ * T_;   // second output

  const int nMean = B_ * G_ * GKV * T_;                // 8,388,608
  zero_f32<<<(nMean + 255) / 256, 256, 0, stream>>>(amean, nMean);

  const int nBlocks = B_ * H_ * (T_ / TT);             // 2048
  tamcad_attn<<<nBlocks, NTHREADS, 0, stream>>>(q, k, v, ki, vi, xout, amean);
}
